// FastSpeech_87024627351674
// MI455X (gfx1250) — matrix operand order repaired and checked by reading, not compiled, not run on hardware
//
#include <hip/hip_runtime.h>

#define NHEAD  8
#define DHEAD  64
#define DMODEL 512
#define SEQ    2048
#define BATCH  4
#define WINDOW 64
#define ROWS   (BATCH * SEQ)          /* 8192 */
#define NQKV   (3 * NHEAD * DHEAD)    /* 1536 */
#define QTILES (SEQ / 16)             /* 128 */

typedef __attribute__((ext_vector_type(16))) _Float16 v16h;
typedef __attribute__((ext_vector_type(8)))  _Float16 v8h;
typedef __attribute__((ext_vector_type(8)))  float    v8f;

static __device__ __forceinline__ v16h cat8(v8h lo, v8h hi) {
  return __builtin_shufflevector(lo, hi, 0,1,2,3,4,5,6,7,8,9,10,11,12,13,14,15);
}

// ---------------------------------------------------------------------------
// Precision staging: f32 -> f16 (and transposes so WMMA B-fragments are
// contiguous per-lane loads).
// ---------------------------------------------------------------------------
__global__ void cvt_f16_kernel(const float* __restrict__ src,
                               _Float16* __restrict__ dst, int n) {
  int i = blockIdx.x * blockDim.x + threadIdx.x;
  if (i < n) dst[i] = (_Float16)src[i];
}

__global__ void transpose_f16_kernel(const float* __restrict__ src,
                                     _Float16* __restrict__ dst,
                                     int rows, int cols) {
  // src[rows][cols] -> dst[cols][rows]
  int i = blockIdx.x * blockDim.x + threadIdx.x;
  if (i < rows * cols) {
    int r = i / cols, c = i % cols;
    dst[c * rows + r] = (_Float16)src[i];
  }
}

// ---------------------------------------------------------------------------
// QKV projection: [8192,512](f16) x [512,1536] -> Q,K,V as [B,H,S,64] f16.
// One wave = one 16x16 output tile, K-loop of 16 x v_wmma_f32_16x16x32_f16.
// ---------------------------------------------------------------------------
__global__ __launch_bounds__(256) void qkv_gemm_kernel(
    const _Float16* __restrict__ A,    // inp f16 [8192,512]
    const _Float16* __restrict__ BT,   // w_qkv^T f16 [1536,512]
    const float*    __restrict__ bias, // [1536]
    _Float16* __restrict__ Q, _Float16* __restrict__ K,
    _Float16* __restrict__ V) {
  const int wid  = threadIdx.x >> 5;
  const int lane = threadIdx.x & 31;
  const int tile = blockIdx.x * 8 + wid;
  const int NT   = NQKV / 16;                 // 96
  const int mt   = tile / NT, nt = tile % NT; // mt < 512
  const int m = lane & 15, hi = lane >> 4;
  const int base8 = hi * 8, base16 = hi * 16;

  const _Float16* arow = A  + (size_t)(mt * 16 + m) * DMODEL;
  const _Float16* brow = BT + (size_t)(nt * 16 + m) * DMODEL;

  v8f c = {};
  for (int kk = 0; kk < DMODEL; kk += 32) {
    v16h a = *(const v16h*)(arow + kk + base16);
    v16h b = *(const v16h*)(brow + kk + base16);
    c = __builtin_amdgcn_wmma_f32_16x16x32_f16(false, a, false, b,
                                               (short)0, c, false, false);
  }

  const int n_g   = nt * 16 + m;
  const float bs  = bias[n_g];
  const int chunk = n_g / (NHEAD * DHEAD);   // uniform across the wave
  const int rem   = n_g % (NHEAD * DHEAD);
  const int h     = rem / DHEAD;             // uniform across the wave
  const int d     = rem % DHEAD;
  for (int r = 0; r < 8; ++r) {
    int row = mt * 16 + r + 8 * hi;
    int bb  = row / SEQ, s = row % SEQ;
    size_t idx = ((size_t)(bb * NHEAD + h) * SEQ + s) * DHEAD + d;
    float v = c[r] + bs;
    if (chunk == 0)      Q[idx] = (_Float16)(v * 0.125f);  // fold 1/sqrt(64)
    else if (chunk == 1) K[idx] = (_Float16)v;
    else                 V[idx] = (_Float16)v;
  }
}

// ---------------------------------------------------------------------------
// Banded flash attention: window |i-j| <= 64. One wave = one (b,h,16-query)
// tile; online softmax over 32-key blocks (pairs of 16-key tiles) so the
// P x V WMMAs use the full 16x16x32 shape (no zero padding, no divergence).
// ---------------------------------------------------------------------------
__global__ __launch_bounds__(256) void attn_kernel(
    const _Float16* __restrict__ Q, const _Float16* __restrict__ K,
    const _Float16* __restrict__ V, _Float16* __restrict__ O /*[B,S,512]*/) {
  __shared__ _Float16 sP [8][16 * 32];   // per-wave P block  [q][k]   (1 KB)
  __shared__ _Float16 sVt[8][64 * 32];   // per-wave V^T block [d][k]  (4 KB)

  const int wid  = threadIdx.x >> 5;
  const int lane = threadIdx.x & 31;
  const int tile = blockIdx.x * 8 + wid;       // 4096 tiles total
  const int b  = tile >> 10;                   // / (NHEAD*QTILES)
  const int h  = (tile >> 7) & 7;
  const int qt = tile & 127;
  const int m = lane & 15, hi = lane >> 4;
  const int base8 = hi * 8, base16 = hi * 16;

  const size_t headoff = (size_t)(b * NHEAD + h) * SEQ * DHEAD;
  const _Float16* Qb = Q + headoff;
  const _Float16* Kb = K + headoff;
  const _Float16* Vb = V + headoff;

  // Q A-fragments for the two 32-wide d-steps (row m, d contiguous)
  const _Float16* qrow = Qb + (size_t)(qt * 16 + m) * DHEAD;
  v16h aq0 = *(const v16h*)(qrow + base16);
  v16h aq1 = *(const v16h*)(qrow + 32 + base16);

  v8f o0 = {}, o1 = {}, o2 = {}, o3 = {};
  float mrun[8], lrun[8];
  for (int r = 0; r < 8; ++r) { mrun[r] = -1e30f; lrun[r] = 0.0f; }

  // 5 blocks of 32 keys covering tiles qt-4 .. qt+5 (tile qt+5 masks to 0)
  for (int p = 0; p < 5; ++p) {
    const int kt0u = qt - 4 + 2 * p;       // first tile of this block
    const int kt1u = kt0u + 1;
    if (kt1u < 0 || kt0u > QTILES - 1) continue;  // block fully off-sequence
    const bool v0 = (kt0u >= 0) && (kt0u <= QTILES - 1);
    const bool v1 = (kt1u >= 0) && (kt1u <= QTILES - 1);
    const int kt0 = v0 ? kt0u : (kt0u < 0 ? 0 : QTILES - 1);
    const int kt1 = v1 ? kt1u : (kt1u < 0 ? 0 : QTILES - 1);

    if (kt1u + 2 <= QTILES - 1) {  // prefetch next block (global_prefetch_b8)
      __builtin_prefetch(Kb + (size_t)((kt0u + 2) * 16 + m) * DHEAD, 0, 1);
      __builtin_prefetch(Vb + (size_t)((kt0u + 2) * 16 + m) * DHEAD, 0, 1);
    }

    // ---- scores for the two 16-key tiles (4 WMMAs, f32 accumulate) ----
    const _Float16* krow0 = Kb + (size_t)(kt0 * 16 + m) * DHEAD;
    const _Float16* krow1 = Kb + (size_t)(kt1 * 16 + m) * DHEAD;
    v8f sc0 = {}, sc1 = {};
    {
      v16h bk0 = *(const v16h*)(krow0 + base16);
      v16h bk1 = *(const v16h*)(krow0 + 32 + base16);
      sc0 = __builtin_amdgcn_wmma_f32_16x16x32_f16(false, aq0, false, bk0, (short)0, sc0, false, false);
      sc0 = __builtin_amdgcn_wmma_f32_16x16x32_f16(false, aq1, false, bk1, (short)0, sc0, false, false);
    }
    {
      v16h bk0 = *(const v16h*)(krow1 + base16);
      v16h bk1 = *(const v16h*)(krow1 + 32 + base16);
      sc1 = __builtin_amdgcn_wmma_f32_16x16x32_f16(false, aq0, false, bk0, (short)0, sc1, false, false);
      sc1 = __builtin_amdgcn_wmma_f32_16x16x32_f16(false, aq1, false, bk1, (short)0, sc1, false, false);
    }

    // ---- stage V^T (32 keys x 64 d) into LDS: sVt[d][k] ----
    {
      const int kts = hi ? kt1 : kt0;      // lane group hi stages tile hi
      const _Float16* vrow = Vb + (size_t)(kts * 16 + m) * DHEAD;
      #pragma unroll
      for (int j = 0; j < 64; ++j) sVt[wid][j * 32 + hi * 16 + m] = vrow[j];
    }

    // ---- window mask + online softmax over 32 columns ----
    float alpha[8];
    const int kg0 = kt0 * 16 + m;
    const int kg1 = kt1 * 16 + m;
    for (int r = 0; r < 8; ++r) {
      int qg = qt * 16 + r + 8 * hi;
      int d0 = qg - kg0; if (d0 < 0) d0 = -d0;
      int d1 = qg - kg1; if (d1 < 0) d1 = -d1;
      float s0 = (!v0 || d0 > WINDOW) ? -1e30f : sc0[r];
      float s1 = (!v1 || d1 > WINDOW) ? -1e30f : sc1[r];
      float v = fmaxf(s0, s1);
      v = fmaxf(v, __shfl_xor(v, 1, 32));
      v = fmaxf(v, __shfl_xor(v, 2, 32));
      v = fmaxf(v, __shfl_xor(v, 4, 32));
      v = fmaxf(v, __shfl_xor(v, 8, 32));
      float mnew = fmaxf(mrun[r], v);
      alpha[r]   = __expf(mrun[r] - mnew);
      float p0   = __expf(s0 - mnew);
      float p1   = __expf(s1 - mnew);
      mrun[r]    = mnew;
      float ps = p0 + p1;
      ps += __shfl_xor(ps, 1, 32);
      ps += __shfl_xor(ps, 2, 32);
      ps += __shfl_xor(ps, 4, 32);
      ps += __shfl_xor(ps, 8, 32);
      lrun[r] = lrun[r] * alpha[r] + ps;
      sP[wid][(r + 8 * hi) * 32 + m]      = (_Float16)p0;
      sP[wid][(r + 8 * hi) * 32 + 16 + m] = (_Float16)p1;
    }
    for (int r = 0; r < 8; ++r) {
      o0[r] *= alpha[r]; o1[r] *= alpha[r]; o2[r] *= alpha[r]; o3[r] *= alpha[r];
    }

    // DS is in-order within a wave; explicit wait before consuming LDS data.
    asm volatile("s_wait_dscnt 0x0" ::: "memory");

    // ---- O += P V with full K=32 fragments ----
    v16h ap = *(const v16h*)&sP[wid][m * 32 + base16];
    v16h bv0 = *(const v16h*)&sVt[wid][(0 * 16 + m) * 32 + base16];
    v16h bv1 = *(const v16h*)&sVt[wid][(1 * 16 + m) * 32 + base16];
    v16h bv2 = *(const v16h*)&sVt[wid][(2 * 16 + m) * 32 + base16];
    v16h bv3 = *(const v16h*)&sVt[wid][(3 * 16 + m) * 32 + base16];
    o0 = __builtin_amdgcn_wmma_f32_16x16x32_f16(false, ap, false, bv0, (short)0, o0, false, false);
    o1 = __builtin_amdgcn_wmma_f32_16x16x32_f16(false, ap, false, bv1, (short)0, o1, false, false);
    o2 = __builtin_amdgcn_wmma_f32_16x16x32_f16(false, ap, false, bv2, (short)0, o2, false, false);
    o3 = __builtin_amdgcn_wmma_f32_16x16x32_f16(false, ap, false, bv3, (short)0, o3, false, false);
  }

  // ---- normalize and store O in [B,S,H*64] layout for the out-projection ----
  for (int r = 0; r < 8; ++r) {
    int s = qt * 16 + r + 8 * hi;
    float inv = 1.0f / lrun[r];
    size_t base = ((size_t)b * SEQ + s) * DMODEL + h * DHEAD;
    O[base +  0 + m] = (_Float16)(o0[r] * inv);
    O[base + 16 + m] = (_Float16)(o1[r] * inv);
    O[base + 32 + m] = (_Float16)(o2[r] * inv);
    O[base + 48 + m] = (_Float16)(o3[r] * inv);
  }
}

// ---------------------------------------------------------------------------
// Output projection + residual + LayerNorm, fused. One block per 16-row
// stripe: 8 waves x 4 N-tiles of WMMA into LDS, then per-row LN.
// ---------------------------------------------------------------------------
__global__ __launch_bounds__(256) void proj_ln_kernel(
    const _Float16* __restrict__ Aattn,  // [8192,512] f16
    const _Float16* __restrict__ WoT,    // [512,512] f16 (transposed)
    const float* __restrict__ inp, const float* __restrict__ g,
    const float* __restrict__ bta, float* __restrict__ out) {
  __shared__ float sX[16][DMODEL];  // 32 KB
  const int wid  = threadIdx.x >> 5;
  const int lane = threadIdx.x & 31;
  const int m = lane & 15, hi = lane >> 4;
  const int base8 = hi * 8, base16 = hi * 16;
  const int mt = blockIdx.x;

  const _Float16* arow = Aattn + (size_t)(mt * 16 + m) * DMODEL;
  for (int t = 0; t < 4; ++t) {
    int nt = wid * 4 + t;
    const _Float16* brow = WoT + (size_t)(nt * 16 + m) * DMODEL;
    v8f c = {};
    for (int kk = 0; kk < DMODEL; kk += 32) {
      v16h a = *(const v16h*)(arow + kk + base16);
      v16h b = *(const v16h*)(brow + kk + base16);
      c = __builtin_amdgcn_wmma_f32_16x16x32_f16(false, a, false, b,
                                                 (short)0, c, false, false);
    }
    int n = nt * 16 + m;
    for (int r = 0; r < 8; ++r) {
      int rl = r + 8 * hi;
      int row = mt * 16 + rl;
      sX[rl][n] = c[r] + inp[(size_t)row * DMODEL + n];  // residual
    }
  }
  __syncthreads();

  for (int rr = 0; rr < 2; ++rr) {
    int rl = wid * 2 + rr;
    int row = mt * 16 + rl;
    float sum = 0.f, sq = 0.f;
    for (int j = lane; j < DMODEL; j += 32) {
      float x = sX[rl][j]; sum += x; sq += x * x;
    }
    for (int msk = 16; msk >= 1; msk >>= 1) {
      sum += __shfl_xor(sum, msk, 32);
      sq  += __shfl_xor(sq,  msk, 32);
    }
    float mu   = sum * (1.0f / DMODEL);
    float var  = sq  * (1.0f / DMODEL) - mu * mu;
    float rstd = rsqrtf(var + 1e-5f);
    for (int j = lane; j < DMODEL; j += 32) {
      out[(size_t)row * DMODEL + j] = (sX[rl][j] - mu) * rstd * g[j] + bta[j];
    }
  }
}

// ---------------------------------------------------------------------------
extern "C" void kernel_launch(void* const* d_in, const int* in_sizes, int n_in,
                              void* d_out, int out_size, void* d_ws,
                              size_t ws_size, hipStream_t stream) {
  (void)in_sizes; (void)n_in; (void)out_size; (void)ws_size;
  const float* inp   = (const float*)d_in[0];
  const float* w_qkv = (const float*)d_in[1];
  const float* b_qkv = (const float*)d_in[2];
  const float* w_o   = (const float*)d_in[3];
  const float* ln_g  = (const float*)d_in[4];
  const float* ln_b  = (const float*)d_in[5];
  float* out = (float*)d_out;

  char* ws = (char*)d_ws;
  size_t off = 0;
  auto carve = [&](size_t bytes) -> void* {
    void* p = ws + off;
    off += (bytes + 255) & ~(size_t)255;
    return p;
  };
  _Float16* inp16 = (_Float16*)carve((size_t)ROWS * DMODEL * 2);
  _Float16* wqkvT = (_Float16*)carve((size_t)NQKV * DMODEL * 2);
  _Float16* woT   = (_Float16*)carve((size_t)DMODEL * DMODEL * 2);
  _Float16* Qh    = (_Float16*)carve((size_t)ROWS * DMODEL * 2);
  _Float16* Kh    = (_Float16*)carve((size_t)ROWS * DMODEL * 2);
  _Float16* Vh    = (_Float16*)carve((size_t)ROWS * DMODEL * 2);
  _Float16* Oh    = (_Float16*)carve((size_t)ROWS * DMODEL * 2);

  cvt_f16_kernel<<<(ROWS * DMODEL + 255) / 256, 256, 0, stream>>>(
      inp, inp16, ROWS * DMODEL);
  transpose_f16_kernel<<<(DMODEL * NQKV + 255) / 256, 256, 0, stream>>>(
      w_qkv, wqkvT, DMODEL, NQKV);
  transpose_f16_kernel<<<(DMODEL * DMODEL + 255) / 256, 256, 0, stream>>>(
      w_o, woT, DMODEL, DMODEL);

  // 512 M-tiles x 96 N-tiles = 49152 waves -> 6144 blocks of 8 waves
  qkv_gemm_kernel<<<(512 * 96) / 8, 256, 0, stream>>>(inp16, wqkvT, b_qkv,
                                                      Qh, Kh, Vh);
  // 4 * 8 * 128 = 4096 attention tiles -> 512 blocks of 8 waves
  attn_kernel<<<4096 / 8, 256, 0, stream>>>(Qh, Kh, Vh, Oh);
  // one block per 16-row stripe
  proj_ln_kernel<<<ROWS / 16, 256, 0, stream>>>(Oh, woT, inp, ln_g, ln_b, out);
}
